// JointGNNEncoder_26980984553612
// MI455X (gfx1250) — matrix operand order repaired and checked by reading, not compiled, not run on hardware
//
#include <hip/hip_runtime.h>
#include <stdint.h>

// ---------------------------------------------------------------------------
// Fused 2-layer GCN encoder for the 24-node hand-joint chain graph.
//   layer1: T = X @ W1                      (WMMA bf16, f32 acc)
//           H = relu(A_norm @ T + b1)       (branch-free sparse 3-term VALU)
//   layer2: U = H @ W2                      (WMMA bf16, f32 acc)
//           out = mean_n relu(A_norm @ U + b2)
// One workgroup = 256 threads = 8 waves processes 2 graphs (48 rows).
// ---------------------------------------------------------------------------

typedef __attribute__((ext_vector_type(16))) __bf16   v16bf;
typedef __attribute__((ext_vector_type(4)))  __bf16   v4bf;
typedef __attribute__((ext_vector_type(8)))  float    v8f;
typedef __attribute__((ext_vector_type(4)))  uint32_t u32x4;

#define N_JOINTS 24
#define GPB      2                   // graphs per block
#define ROWS     (GPB * N_JOINTS)    // 48 rows -> 3 M-tiles of 16
#define CIN      64
#define HDIM     256
#define XSTR     72                  // padded LDS row stride (elems), 16B aligned
#define TSTR     264                 // padded LDS row stride for T / H (elems)

// Kinematic-chain neighbor lists (fixed graph structure; -1 = none).
__constant__ int NB0[N_JOINTS] = { 1, 0, 3, 2, 3, 4, 7, 6, 7, 8,11,10,11,12,15,14,15,16,17,20,19,20,21,22};
__constant__ int NB1[N_JOINTS] = {-1,-1,-1, 4, 5,-1,-1, 8, 9,-1,-1,12,13,-1,-1,16,17,18,-1,-1,21,22,23,-1};

union ABFrag { v16bf v; u32x4 q[2]; };

// A-matrix fragment (16x32 bf16) from LDS, ISA layout:
//   lane m (row = mt*16 + (lane&15)), K group base (lane>>4)*8, +16 for v[4:7].
__device__ __forceinline__ ABFrag loadA_lds(const __bf16* base, int stride,
                                            int mt, int kt, int lane) {
    int row = mt * 16 + (lane & 15);
    int kb  = kt * 32 + ((lane >> 4) * 8);
    const u32x4* p = (const u32x4*)(base + row * stride + kb);
    ABFrag r; r.q[0] = p[0]; r.q[1] = p[2];     // second half at +16 elements
    return r;
}

// B-matrix fragment (32x16 bf16) from global W^T (row = out col, K contiguous):
//   lane n (col = ct*16 + (lane&15)), K group base (lane>>4)*8, +16 for v[4:7].
__device__ __forceinline__ ABFrag loadB_glob(const __bf16* WT, int K,
                                             int ct, int kt, int lane) {
    int col = ct * 16 + (lane & 15);
    int kb  = kt * 32 + ((lane >> 4) * 8);
    const u32x4* p = (const u32x4*)(WT + col * K + kb);
    ABFrag r; r.q[0] = p[0]; r.q[1] = p[2];
    return r;
}

// C/D layout: VGPR r holds (M = mbase + r, N = col), mbase = (lane>>4)*8.
__device__ __forceinline__ void storeC_bf16(__bf16* T, int stride,
                                            int mt, int ct, int lane, v8f acc) {
    int col   = ct * 16 + (lane & 15);
    int mbase = mt * 16 + ((lane >> 4) * 8);
#pragma unroll
    for (int r = 0; r < 8; ++r)
        T[(mbase + r) * stride + col] = (__bf16)acc[r];
}

__device__ __forceinline__ v8f wmma_bf16(ABFrag a, ABFrag b, v8f c) {
    return __builtin_amdgcn_wmma_f32_16x16x32_bf16(
        /*neg_a=*/false, a.v, /*neg_b=*/false, b.v,
        /*c_mod=*/(short)0, c, /*reuse_a=*/false, /*reuse_b=*/false);
}

// ---------------------------------------------------------------------------
// Kernel 0: one-time f32 -> bf16 transposed weight conversion into d_ws.
//   W1 (64x256)  -> W1T (256x64)   at ws elem offset 0
//   W2 (256x256) -> W2T (256x256)  at ws elem offset 16384
// ---------------------------------------------------------------------------
__global__ void gcn_convert_weights(const float* __restrict__ W1,
                                    const float* __restrict__ W2,
                                    __bf16* __restrict__ W1T,
                                    __bf16* __restrict__ W2T) {
    int i = blockIdx.x * blockDim.x + threadIdx.x;
    if (i < CIN * HDIM) {                       // 16384
        int n = i >> 6, k = i & 63;
        W1T[n * CIN + k] = (__bf16)W1[k * HDIM + n];
    } else if (i < CIN * HDIM + HDIM * HDIM) {  // + 65536
        int t = i - CIN * HDIM;
        int n = t >> 8, k = t & 255;
        W2T[n * HDIM + k] = (__bf16)W2[k * HDIM + n];
    }
}

// ---------------------------------------------------------------------------
// Kernel 1: fused GCN, one block per pair of graphs.
// ---------------------------------------------------------------------------
__global__ __launch_bounds__(256)
void gcn_fused(const float* __restrict__ feats,   // (B, 24, 64)
               const float* __restrict__ b1,      // (256)
               const float* __restrict__ b2,      // (256)
               const float* __restrict__ Anorm,   // (24, 24)
               const __bf16* __restrict__ W1T,
               const __bf16* __restrict__ W2T,
               float* __restrict__ out) {         // (B, 256)
    __shared__ __attribute__((aligned(16))) __bf16 Xs[ROWS * XSTR];
    __shared__ __attribute__((aligned(16))) __bf16 Ts[ROWS * TSTR]; // T, then U
    __shared__ __attribute__((aligned(16))) __bf16 Hs[ROWS * TSTR];
    // Branch-free sparse-aggregation tables: row n of A_norm reduced to
    // self weight + two (index, weight) pairs (weight 0 for missing edge).
    __shared__ float sw[N_JOINTS], w0s[N_JOINTS], w1s[N_JOINTS];
    __shared__ int   i0s[N_JOINTS], i1s[N_JOINTS];
    __shared__ float b1s[HDIM], b2s[HDIM];

    const int tid  = threadIdx.x;
    const int lane = tid & 31;
    const int wave = tid >> 5;          // 0..7 -> owns col-tiles {2w, 2w+1}
    const int pair = blockIdx.x;        // graphs {2*pair, 2*pair+1}
    const long gbase = (long)pair * GPB * N_JOINTS * CIN;

    // ---- stage aggregation tables + biases -------------------------------
    if (tid < N_JOINTS) {
        const int n  = tid;
        const int a0 = NB0[n], a1 = NB1[n];
        sw[n]  = Anorm[n * N_JOINTS + n];
        i0s[n] = a0;
        w0s[n] = Anorm[n * N_JOINTS + a0];
        i1s[n] = (a1 >= 0) ? a1 : n;
        w1s[n] = (a1 >= 0) ? Anorm[n * N_JOINTS + a1] : 0.0f;
    }
    if (tid < HDIM) { b1s[tid] = b1[tid]; b2s[tid] = b2[tid]; }

    // ---- stage X (f32 -> bf16) into LDS: float4 loads, 4xbf16 stores -----
    {
        const float4* src = (const float4*)(feats + gbase);
#pragma unroll
        for (int i = tid; i < ROWS * CIN / 4; i += 256) {   // 3 iterations
            float4 v = src[i];
            int e   = i * 4;
            int row = e >> 6, k = e & 63;
            v4bf p = { (__bf16)v.x, (__bf16)v.y, (__bf16)v.z, (__bf16)v.w };
            *(v4bf*)(Xs + row * XSTR + k) = p;
        }
    }
    __syncthreads();

    // ---- layer 1 GEMM: T = X @ W1  (K = 64 -> 2 WMMAs per tile) ----------
#pragma unroll
    for (int mt = 0; mt < 3; ++mt) {
        ABFrag a0 = loadA_lds(Xs, XSTR, mt, 0, lane);
        ABFrag a1 = loadA_lds(Xs, XSTR, mt, 1, lane);
#pragma unroll
        for (int cto = 0; cto < 2; ++cto) {
            const int ct = wave * 2 + cto;
            v8f acc = {};
            acc = wmma_bf16(a0, loadB_glob(W1T, CIN, ct, 0, lane), acc);
            acc = wmma_bf16(a1, loadB_glob(W1T, CIN, ct, 1, lane), acc);
            storeC_bf16(Ts, TSTR, mt, ct, lane, acc);
        }
    }
    __syncthreads();

    // ---- aggregation 1: H = relu(A @ T + b1), branch-free 3-term ---------
    for (int idx = tid; idx < ROWS * HDIM; idx += 256) {
        int row = idx >> 8, c = idx & 255;
        int g = (row >= N_JOINTS) ? 1 : 0;
        int base = g * N_JOINTS, n = row - base;
        float acc = sw[n]  * (float)Ts[(base + n)        * TSTR + c]
                  + w0s[n] * (float)Ts[(base + i0s[n])   * TSTR + c]
                  + w1s[n] * (float)Ts[(base + i1s[n])   * TSTR + c]
                  + b1s[c];
        Hs[row * TSTR + c] = (__bf16)fmaxf(acc, 0.0f);
    }
    __syncthreads();

    // ---- layer 2 GEMM: U = H @ W2  (K = 256 -> 8 WMMAs per tile) ---------
    // Row-tile A-fragments loaded once, reused for both column-tiles.
#pragma unroll
    for (int mt = 0; mt < 3; ++mt) {
        ABFrag a[8];
#pragma unroll
        for (int kt = 0; kt < 8; ++kt)
            a[kt] = loadA_lds(Hs, TSTR, mt, kt, lane);
#pragma unroll
        for (int cto = 0; cto < 2; ++cto) {
            const int ct = wave * 2 + cto;
            v8f acc = {};
#pragma unroll
            for (int kt = 0; kt < 8; ++kt)
                acc = wmma_bf16(a[kt], loadB_glob(W2T, HDIM, ct, kt, lane), acc);
            storeC_bf16(Ts, TSTR, mt, ct, lane, acc);    // reuse Ts for U
        }
    }
    __syncthreads();

    // ---- aggregation 2 + bias + relu + mean pool -------------------------
    for (int idx = tid; idx < GPB * HDIM; idx += 256) {   // 2 iterations
        int g = idx >> 8, c = idx & 255;
        const float bias = b2s[c];
        float s = 0.0f;
#pragma unroll
        for (int n = 0; n < N_JOINTS; ++n) {
            float acc = sw[n]  * (float)Ts[(g * N_JOINTS + n)        * TSTR + c]
                      + w0s[n] * (float)Ts[(g * N_JOINTS + i0s[n])   * TSTR + c]
                      + w1s[n] * (float)Ts[(g * N_JOINTS + i1s[n])   * TSTR + c]
                      + bias;
            s += fmaxf(acc, 0.0f);
        }
        out[((long)pair * GPB + g) * HDIM + c] = s * (1.0f / 24.0f);
    }
}

// ---------------------------------------------------------------------------
extern "C" void kernel_launch(void* const* d_in, const int* in_sizes, int n_in,
                              void* d_out, int out_size, void* d_ws, size_t ws_size,
                              hipStream_t stream) {
    const float* feats = (const float*)d_in[0];   // (8192, 24, 64)
    const float* W1    = (const float*)d_in[1];   // (64, 256)
    const float* b1    = (const float*)d_in[2];   // (256)
    const float* W2    = (const float*)d_in[3];   // (256, 256)
    const float* b2    = (const float*)d_in[4];   // (256)
    const float* An    = (const float*)d_in[5];   // (24, 24)
    float* out         = (float*)d_out;

    const int B = in_sizes[0] / (N_JOINTS * CIN); // 8192

    __bf16* W1T = (__bf16*)d_ws;                  // 256x64
    __bf16* W2T = (__bf16*)d_ws + CIN * HDIM;     // 256x256

    // One-time weight transpose + bf16 convert (re-done every call: deterministic).
    const int wtot = CIN * HDIM + HDIM * HDIM;    // 81920
    gcn_convert_weights<<<(wtot + 255) / 256, 256, 0, stream>>>(W1, W2, W1T, W2T);

    // Fused GCN: one block per 2 graphs.
    gcn_fused<<<B / GPB, 256, 0, stream>>>(feats, b1, b2, An, W1T, W2T, out);
}
